// StackedSENEGNN_37168646980028
// MI455X (gfx1250) — matrix operand order repaired and checked by reading, not compiled, not run on hardware
//
#include <hip/hip_runtime.h>
#include <stdint.h>

#define NN   50000
#define NE   800000
#define DIMD 128
#define CDIM 3
#define NL   3
#define MT   4          // M row-tiles per block (64 rows), B held in registers

typedef __attribute__((ext_vector_type(16))) __bf16 v16bf;
typedef __attribute__((ext_vector_type(8)))  float  v8f;
typedef __attribute__((ext_vector_type(4)))  float  f32x4;

__device__ __forceinline__ unsigned short f2bf(float f) {
  union { float f; unsigned int u; } v; v.f = f;
  unsigned int u = v.u;
  u += 0x7FFFu + ((u >> 16) & 1u);   // round-to-nearest-even
  return (unsigned short)(u >> 16);
}

// ---------------- fp32 -> bf16 convert ----------------
__global__ void k_f32_to_bf16(const float* __restrict__ src,
                              unsigned short* __restrict__ dst, int n) {
  int i = blockIdx.x * blockDim.x + threadIdx.x;
  if (i < n) dst[i] = f2bf(src[i]);
}

// ------------- transpose one DxD weight to bf16 (WT[n][k] = W[k][n]) -------------
__global__ void k_transpose_bf16(const float* __restrict__ W,
                                 unsigned short* __restrict__ WT) {
  int i = blockIdx.x * blockDim.x + threadIdx.x;   // over DIMD*DIMD
  int k = i >> 7, n = i & 127;
  WT[n * DIMD + k] = f2bf(W[k * DIMD + n]);
}

// ---------------- zero fill ----------------
__global__ void k_zero(float* __restrict__ p, int n) {
  int i = blockIdx.x * blockDim.x + threadIdx.x;
  if (i < n) p[i] = 0.0f;
}

// ---------------- WMMA bf16 GEMM:  C = A(rows x 128) * B(128 x 128) + bias ----------------
// A row-major bf16; BT is B transposed (BT[n][k]) bf16 so the B operand loads contiguously.
// Block = 256 threads = 8 waves. Wave w owns cols [16w, 16w+16) and holds its full B
// stripe (16 cols x K=128 = 32 VGPRs) in registers, then sweeps MT=4 row tiles
// (64 rows per block), 4 WMMAs each -> 16 v_wmma per wave, 4x less B traffic.
__global__ void __launch_bounds__(256)
k_gemm_bias(const unsigned short* __restrict__ A,
            const unsigned short* __restrict__ BT,
            const float* __restrict__ bias,
            float* __restrict__ C, int rows) {
  const int lane  = threadIdx.x & 31;
  const int wave  = threadIdx.x >> 5;
  const int n0    = wave * 16;
  const int rbase = blockIdx.x * (16 * MT);
  const int half  = lane >> 4;     // lane group 0/1
  const int l16   = lane & 15;

  // B 32x16 bf16 operand, k order used here (same as A): lane -> col n0+l16; lanes 0-15 K=kk+0..15,
  // lanes 16-31 K=kk+16..31, contiguous in transposed storage.
  const unsigned short* bptr = BT + (size_t)(n0 + l16) * DIMD + half * 16;
  union { v16bf v; f32x4 q[2]; } b[4];
  #pragma unroll
  for (int c = 0; c < 4; ++c) {
    b[c].q[0] = *(const f32x4*)(bptr + c * 32);
    b[c].q[1] = *(const f32x4*)(bptr + c * 32 + 8);
  }

  // speculative prefetch of the next block's first A stripe (global_prefetch_b8)
  __builtin_prefetch((const void*)(A + (size_t)(rbase + 16 * MT + l16) * DIMD), 0, 0);

  #pragma unroll
  for (int ti = 0; ti < MT; ++ti) {
    const int r0 = rbase + ti * 16;
    if (r0 >= rows) break;                 // wave-uniform: EXEC stays all-1s for WMMA

    const unsigned short* aptr = A + (size_t)(r0 + l16) * DIMD;
    v8f acc = {};
    #pragma unroll
    for (int c = 0; c < 4; ++c) {
      union { v16bf v; f32x4 q[2]; } a;
      // A 16x32 bf16, k order used here (same as B): lanes 0-15 K = kk+0..15; lanes 16-31 K = kk+16..31
      a.q[0] = *(const f32x4*)(aptr + c * 32 + half * 16);
      a.q[1] = *(const f32x4*)(aptr + c * 32 + half * 16 + 8);
      acc = __builtin_amdgcn_wmma_f32_16x16x32_bf16(false, a.v, false, b[c].v,
                                                    (short)0, acc, false, false);
    }

    const int   ccol = n0 + l16;
    const float bv   = bias[ccol];
    #pragma unroll
    for (int v = 0; v < 8; ++v) {
      const int crow = r0 + half * 8 + v;  // C/D layout: VGPR v -> M = v + 8*(lane/16)
      C[(size_t)crow * DIMD + ccol] = acc[v] + bv;
    }
  }
}

// ---------------- edge scatter-add: out[dst] += h[src], one wave32 per edge ----------------
__global__ void __launch_bounds__(256)
k_scatter(const float* __restrict__ h, const int* __restrict__ src,
          const int* __restrict__ dst, float* __restrict__ out, int nedges) {
  const int lane = threadIdx.x & 31;
  const int e    = blockIdx.x * 8 + (threadIdx.x >> 5);
  if (e >= nedges) return;
  const int s = src[e], d = dst[e];
  const f32x4 v = *(const f32x4*)(h + (size_t)s * DIMD + lane * 4);
  float* o = out + (size_t)d * DIMD + lane * 4;
  #pragma unroll
  for (int j = 0; j < 4; ++j)
    (void)__hip_atomic_fetch_add(o + j, v[j], __ATOMIC_RELAXED, __HIP_MEMORY_SCOPE_AGENT);
}

// ---------------- per-node: pos/R/t update (incl. 3x3 QR) + bf16 re-encode of x ----------------
__global__ void k_node_update(const float* __restrict__ agg,
                              const float* __restrict__ Wc, const float* __restrict__ bc,
                              const float* __restrict__ WR, const float* __restrict__ bR,
                              const float* __restrict__ Wt, const float* __restrict__ bt,
                              const float* __restrict__ pos_in,
                              float* __restrict__ pos, float* __restrict__ R,
                              float* __restrict__ t,
                              unsigned short* __restrict__ xbf, int first) {
  const int i = blockIdx.x * blockDim.x + threadIdx.x;
  if (i >= NN) return;
  const float* o = agg + (size_t)i * DIMD;

  float dc = 0.0f;
  float dR[9] = {0, 0, 0, 0, 0, 0, 0, 0, 0};
  float dt[3] = {0, 0, 0};
  for (int k = 0; k < DIMD; ++k) {
    const float ov = o[k];
    dc += ov * Wc[k];
    #pragma unroll
    for (int j = 0; j < 9; ++j) dR[j] += ov * WR[k * 9 + j];
    #pragma unroll
    for (int j = 0; j < 3; ++j) dt[j] += ov * Wt[k * 3 + j];
    xbf[(size_t)i * DIMD + k] = f2bf(ov);     // x for next layer's WMMA GEMM
  }
  dc += bc[0];
  #pragma unroll
  for (int j = 0; j < 9; ++j) dR[j] += bR[j];

  // pos += (out @ Wc + bc) broadcast over 3 coords
  #pragma unroll
  for (int j = 0; j < 3; ++j) {
    const float p = first ? pos_in[(size_t)i * 3 + j] : pos[(size_t)i * 3 + j];
    pos[(size_t)i * 3 + j] = p + dc;
  }
  // t += out @ Wt + bt
  #pragma unroll
  for (int j = 0; j < 3; ++j) {
    const float tv = first ? 0.0f : t[(size_t)i * 3 + j];
    t[(size_t)i * 3 + j] = tv + dt[j] + bt[j];
  }

  // QR (modified Gram-Schmidt on columns of dR)
  float q[9];
  float n0 = rsqrtf(dR[0]*dR[0] + dR[3]*dR[3] + dR[6]*dR[6] + 1e-12f);
  q[0] = dR[0]*n0; q[3] = dR[3]*n0; q[6] = dR[6]*n0;
  float d01 = q[0]*dR[1] + q[3]*dR[4] + q[6]*dR[7];
  float c1x = dR[1] - d01*q[0], c1y = dR[4] - d01*q[3], c1z = dR[7] - d01*q[6];
  float n1 = rsqrtf(c1x*c1x + c1y*c1y + c1z*c1z + 1e-12f);
  q[1] = c1x*n1; q[4] = c1y*n1; q[7] = c1z*n1;
  float d02 = q[0]*dR[2] + q[3]*dR[5] + q[6]*dR[8];
  float d12 = q[1]*dR[2] + q[4]*dR[5] + q[7]*dR[8];
  float c2x = dR[2] - d02*q[0] - d12*q[1];
  float c2y = dR[5] - d02*q[3] - d12*q[4];
  float c2z = dR[8] - d02*q[6] - d12*q[7];
  float n2 = rsqrtf(c2x*c2x + c2y*c2y + c2z*c2z + 1e-12f);
  q[2] = c2x*n2; q[5] = c2y*n2; q[8] = c2z*n2;

  // R = Q @ R_prev  (R_prev = I on first layer)
  float Rp[9], Rn[9];
  #pragma unroll
  for (int j = 0; j < 9; ++j)
    Rp[j] = first ? ((j % 4 == 0) ? 1.0f : 0.0f) : R[(size_t)i * 9 + j];
  #pragma unroll
  for (int r = 0; r < 3; ++r)
    #pragma unroll
    for (int c = 0; c < 3; ++c)
      Rn[r*3 + c] = q[r*3+0]*Rp[0*3+c] + q[r*3+1]*Rp[1*3+c] + q[r*3+2]*Rp[2*3+c];
  #pragma unroll
  for (int j = 0; j < 9; ++j) R[(size_t)i * 9 + j] = Rn[j];
}

extern "C" void kernel_launch(void* const* d_in, const int* in_sizes, int n_in,
                              void* d_out, int out_size, void* d_ws, size_t ws_size,
                              hipStream_t stream) {
  (void)in_sizes; (void)n_in; (void)out_size; (void)ws_size;

  const float* x      = (const float*)d_in[0];
  const float* pos_in = (const float*)d_in[1];
  const int*   ei     = (const int*)  d_in[2];
  const float* Wl     = (const float*)d_in[3];
  const float* bl     = (const float*)d_in[4];
  const float* Wc     = (const float*)d_in[5];
  const float* bc     = (const float*)d_in[6];
  const float* WR     = (const float*)d_in[7];
  const float* bR     = (const float*)d_in[8];
  const float* Wt     = (const float*)d_in[9];
  const float* bt     = (const float*)d_in[10];
  const float* Wf     = (const float*)d_in[11];
  const float* bf     = (const float*)d_in[12];

  float* z   = (float*)d_out;                       // (N, D)
  float* pos = z   + (size_t)NN * DIMD;             // (N, 3)
  float* R   = pos + (size_t)NN * CDIM;             // (N, 3, 3)
  float* t   = R   + (size_t)NN * CDIM * CDIM;      // (N, 3)

  // workspace carve-up
  size_t off = 0;
  char*  ws  = (char*)d_ws;
  auto bump = [&](size_t bytes) -> char* {
    char* p = ws + off;
    off += (bytes + 255) & ~(size_t)255;
    return p;
  };
  unsigned short* xbf = (unsigned short*)bump((size_t)NN * DIMD * 2);
  float*          h   = (float*)         bump((size_t)NN * DIMD * 4);
  float*          agg = (float*)         bump((size_t)NN * DIMD * 4);
  unsigned short* WlT = (unsigned short*)bump((size_t)NL * DIMD * DIMD * 2);
  unsigned short* WfT = (unsigned short*)bump((size_t)DIMD * DIMD * 2);

  const int* src = ei;
  const int* dst = ei + NE;

  const int gemm_grid = (NN + 16 * MT - 1) / (16 * MT);

  // precision staging: x and weights -> bf16 (weights transposed for WMMA B layout)
  k_f32_to_bf16<<<(NN * DIMD + 255) / 256, 256, 0, stream>>>(x, xbf, NN * DIMD);
  for (int i = 0; i < NL; ++i)
    k_transpose_bf16<<<(DIMD * DIMD) / 256, 256, 0, stream>>>(Wl + (size_t)i * DIMD * DIMD,
                                                              WlT + (size_t)i * DIMD * DIMD);
  k_transpose_bf16<<<(DIMD * DIMD) / 256, 256, 0, stream>>>(Wf, WfT);

  for (int i = 0; i < NL; ++i) {
    k_gemm_bias<<<gemm_grid, 256, 0, stream>>>(xbf, WlT + (size_t)i * DIMD * DIMD,
                                               bl + (size_t)i * DIMD, h, NN);
    k_zero<<<(NN * DIMD + 255) / 256, 256, 0, stream>>>(agg, NN * DIMD);
    k_scatter<<<(NE + 7) / 8, 256, 0, stream>>>(h, src, dst, agg, NE);
    k_node_update<<<(NN + 255) / 256, 256, 0, stream>>>(
        agg,
        Wc + (size_t)i * DIMD, bc + i,
        WR + (size_t)i * DIMD * 9, bR + (size_t)i * 9,
        Wt + (size_t)i * DIMD * 3, bt + (size_t)i * 3,
        pos_in, pos, R, t, xbf, (i == 0) ? 1 : 0);
  }

  // final fc: z = x @ Wf + bf
  k_gemm_bias<<<gemm_grid, 256, 0, stream>>>(xbf, WfT, bf, z, NN);
}
